// Attention_137438953785
// MI455X (gfx1250) — matrix operand order repaired and checked by reading, not compiled, not run on hardware
//
#include <hip/hip_runtime.h>

// ---------------- problem constants ----------------
constexpr int E  = 128;          // head dim / embed
constexpr int H  = 8;            // heads
constexpr int B  = 2;
constexpr int T  = 4096;
constexpr int HE = H * E;        // 1024
constexpr int M  = B * T;        // 8192 rows

typedef __attribute__((ext_vector_type(8)))  float  v8f;
typedef __attribute__((ext_vector_type(8)))  __bf16 v8bf;
typedef __attribute__((ext_vector_type(16))) __bf16 v16bf;

// --------- CDNA5 async global->LDS path (probe via __has_builtin) ----------
#if defined(__has_builtin)
#if __has_builtin(__builtin_amdgcn_global_load_async_to_lds_b128)
#define HAVE_ASYNC_LDS 1
#endif
#endif
#ifndef HAVE_ASYNC_LDS
#define HAVE_ASYNC_LDS 0
#endif

#if HAVE_ASYNC_LDS
// Builtin param 1 is 'int __vector_size(16) __device__*' (AS1 int4*).  Build
// AS1/AS3 int4 pointers via integer casts (flat-LDS addr[31:0] == LDS offset).
typedef int vsi4 __attribute__((vector_size(16)));
typedef vsi4 __attribute__((address_space(1)))* as1_i4p;
typedef vsi4 __attribute__((address_space(3)))* as3_i4p;

__device__ inline void async_copy16(const __bf16* g, __bf16* l) {
    __builtin_amdgcn_global_load_async_to_lds_b128(
        (as1_i4p)(unsigned long long)(uintptr_t)g,
        (as3_i4p)(unsigned int)(uintptr_t)l, 0, 0);
    __builtin_amdgcn_global_load_async_to_lds_b128(
        (as1_i4p)((unsigned long long)(uintptr_t)g + 16ull),
        (as3_i4p)(unsigned int)((uintptr_t)l + 16u), 0, 0);
}
__device__ inline void wait_async0() {
#if __has_builtin(__builtin_amdgcn_s_wait_asynccnt)
    __builtin_amdgcn_s_wait_asynccnt(0);
#else
    asm volatile("s_wait_asynccnt 0" ::: "memory");
#endif
}
#endif

// ---------------- DPP16 butterfly reductions (no LDS traffic) ----------------
// Reduction is within 16-lane halves == one DPP row.  Steps: xor1 (quad_perm
// [1,0,3,2]), xor2 (quad_perm [2,3,0,1]), row_half_mirror, row_mirror.
template <int CTRL>
__device__ inline float dpp_xfer(float x) {
    return __builtin_bit_cast(float,
        __builtin_amdgcn_update_dpp(0, __builtin_bit_cast(int, x),
                                    CTRL, 0xF, 0xF, true));
}
__device__ inline float red16_max(float x) {
    x = fmaxf(x, dpp_xfer<0xB1>(x));    // quad_perm [1,0,3,2]
    x = fmaxf(x, dpp_xfer<0x4E>(x));    // quad_perm [2,3,0,1]
    x = fmaxf(x, dpp_xfer<0x141>(x));   // row_half_mirror
    x = fmaxf(x, dpp_xfer<0x140>(x));   // row_mirror
    return x;
}
__device__ inline float red16_sum(float x) {
    x += dpp_xfer<0xB1>(x);
    x += dpp_xfer<0x4E>(x);
    x += dpp_xfer<0x141>(x);
    x += dpp_xfer<0x140>(x);
    return x;
}

// ---------------- WMMA helpers ----------------
__device__ inline v8f wmma_bf16(v16bf a, v16bf b, v8f c) {
    return __builtin_amdgcn_wmma_f32_16x16x32_bf16(false, a, false, b,
                                                   (short)0, c, false, false);
}

__device__ inline v16bf cat16(v8bf lo, v8bf hi) {
    return __builtin_shufflevector(lo, hi, 0, 1, 2, 3, 4, 5, 6, 7,
                                           8, 9, 10, 11, 12, 13, 14, 15);
}

// A-fragment 16x32 bf16 (MxK) from row-major storage (stride in elements).
__device__ inline v16bf load_a(const __bf16* base, long stride, int mbase,
                               int kbase, int lane) {
    const int hl = (lane >> 4) & 1;
    const int m  = lane & 15;
    const __bf16* p = base + (long)(mbase + m) * stride + kbase + hl * 8;
    v8bf lo = *(const v8bf*)(p);
    v8bf hi = *(const v8bf*)(p + 16);
    return cat16(lo, hi);
}

// B-fragment 32x16 bf16 (KxN) from transposed [N,K] storage (row n contiguous in K).
__device__ inline v16bf load_b(const __bf16* baseT, long stride, int nbase,
                               int kbase, int lane) {
    const int hl = (lane >> 4) & 1;
    const int n  = lane & 15;
    const __bf16* p = baseT + (long)(nbase + n) * stride + kbase + hl * 8;
    v8bf lo = *(const v8bf*)(p);
    v8bf hi = *(const v8bf*)(p + 16);
    return cat16(lo, hi);
}

// ---------------- conversion kernels ----------------
__global__ void cvt_bf16_kernel(const float* __restrict__ src,
                                __bf16* __restrict__ dst, int n, float scale) {
    int i = blockIdx.x * blockDim.x + threadIdx.x;
    if (i < n) dst[i] = (__bf16)(src[i] * scale);
}

__global__ void tcvt_bf16_kernel(const float* __restrict__ src,
                                 __bf16* __restrict__ dst, int rows, int cols) {
    int i = blockIdx.x * blockDim.x + threadIdx.x;
    if (i < rows * cols) {
        int r = i / cols, c = i % cols;
        dst[(long)c * rows + r] = (__bf16)src[i];
    }
}

// ---------------- projection GEMM: X[M,E] @ W[E,HE] (Wt stored [HE,E]) ----------------
template <int MODE>
__global__ __launch_bounds__(128) void proj_gemm_kernel(
        const __bf16* __restrict__ X, const __bf16* __restrict__ Wt,
        __bf16* __restrict__ Y) {
    const int lane = threadIdx.x & 31;
    const int hl = lane >> 4, n = lane & 15;
    const int gw = blockIdx.x * 4 + (threadIdx.x >> 5);
    const int mt = gw >> 6;
    const int nt = gw & 63;

    v8f acc = {};
#pragma unroll
    for (int kk = 0; kk < E; kk += 32) {
        v16bf a = load_a(X, E, mt * 16, kk, lane);
        v16bf b = load_b(Wt, E, nt * 16, kk, lane);
        acc = wmma_bf16(a, b, acc);
    }
#pragma unroll
    for (int r = 0; r < 8; ++r) {
        const int mrow = mt * 16 + r + 8 * hl;
        const int col  = nt * 16 + n;
        if constexpr (MODE == 0) {
            Y[(long)mrow * HE + col] = (__bf16)acc[r];
        } else {
            const int bb = mrow >> 12, t = mrow & (T - 1);
            const int hh = col >> 7,  e = col & (E - 1);
            Y[((long)(bb * H + hh) * E + e) * T + t] = (__bf16)acc[r];
        }
    }
}

// ---------------- flash attention ----------------
// 256 threads = 8 waves, all on one (b,h); block owns 128 consecutive query rows.
// K/V tiles (32 keys) staged once per block into double-buffered LDS via async
// DMA, shared by all 8 waves -> 8x reduction in L2-side K/V traffic.
__global__ __launch_bounds__(256) void flash_attn_kernel(
        const __bf16* __restrict__ Qp,   // [M, HE] bf16 (Q pre-scaled by 1/sqrt(E))
        const __bf16* __restrict__ Kp,   // [M, HE] bf16
        const __bf16* __restrict__ Vt,   // [B, H, E, T] bf16
        __bf16* __restrict__ Op) {       // [M, HE] bf16
    constexpr int KSTR = 136;            // padded stride: conflict-free ds_load_b128
    constexpr int VSTR = 40;
    constexpr int PSTR = 40;
    constexpr int NT   = T / 32;         // 128 key tiles

    __shared__ __align__(16) __bf16 Kl[2][32 * KSTR];
    __shared__ __align__(16) __bf16 Vl[2][128 * VSTR];
    __shared__ __align__(16) __bf16 Pl[8][16 * PSTR];

    const int tid  = threadIdx.x;
    const int lane = tid & 31;
    const int wave = tid >> 5;
    const int hl = lane >> 4, n = lane & 15;

    const int qblk = blockIdx.x & 31;    // 32 query blocks of 128 rows per (b,h)
    const int bh   = blockIdx.x >> 5;
    const int h = bh & (H - 1), b = bh >> 3;
    const int t0 = qblk * 128 + wave * 16;

    const __bf16* Qbase = Qp + (long)b * T * HE + h * E;
    const __bf16* Kbase = Kp + (long)b * T * HE + h * E;
    const __bf16* Vbase = Vt + (long)(b * H + h) * E * T;

    // per-thread staging slice: 16 elems of K, 16 elems of V per tile
    const int kr = tid >> 3, kc = (tid & 7) * 16;   // K tile: 32 x 128
    const int vr = tid >> 1, vc = (tid & 1) * 16;   // V tile: 128 x 32
    const int okK = kr * KSTR + kc;
    const int okV = vr * VSTR + vc;
    const __bf16* gK = Kbase + (long)kr * HE + kc;  // tile kt source: + kt*32*HE
    const __bf16* gV = Vbase + (long)vr * T + vc;   // tile kt source: + kt*32

    // Q fragments held in registers for the whole pass
    v16bf qa[4];
#pragma unroll
    for (int kk = 0; kk < 4; ++kk) qa[kk] = load_a(Qbase, HE, t0, kk * 32, lane);

    v8f o[8];
#pragma unroll
    for (int e = 0; e < 8; ++e) o[e] = {};
    float mrow[8], lrow[8];
#pragma unroll
    for (int r = 0; r < 8; ++r) { mrow[r] = -1e30f; lrow[r] = 0.0f; }

    __bf16* pl = &Pl[wave][0];

#if HAVE_ASYNC_LDS
    // prologue: async-stage tile 0 into buffer 0
    async_copy16(gK, &Kl[0][okK]);
    async_copy16(gV, &Vl[0][okV]);
    const __bf16* gKn = gK + (long)32 * HE;
    const __bf16* gVn = gV + 32;
    wait_async0();
    __syncthreads();
#else
    // prologue: prefetch tile 0 into registers
    v8bf rk0 = *(const v8bf*)gK, rk1 = *(const v8bf*)(gK + 8);
    v8bf rv0 = *(const v8bf*)gV, rv1 = *(const v8bf*)(gV + 8);
    const __bf16* gKn = gK + (long)32 * HE;
    const __bf16* gVn = gV + 32;
#endif

    for (int kt = 0; kt < NT; ++kt) {
        const int cur = kt & 1;

#if HAVE_ASYNC_LDS
        if (kt + 1 < NT) {                 // overlap next tile's DMA with compute
            async_copy16(gKn, &Kl[1 - cur][okK]);
            async_copy16(gVn, &Vl[1 - cur][okV]);
            gKn += (long)32 * HE;
            gVn += 32;
        }
#else
        // commit prefetched regs to current LDS buffer, then barrier
        *(v8bf*)(&Kl[cur][okK])     = rk0;
        *(v8bf*)(&Kl[cur][okK + 8]) = rk1;
        *(v8bf*)(&Vl[cur][okV])     = rv0;
        *(v8bf*)(&Vl[cur][okV + 8]) = rv1;
        __syncthreads();
        if (kt + 1 < NT) {                 // prefetch next tile, overlapped
            rk0 = *(const v8bf*)gKn; rk1 = *(const v8bf*)(gKn + 8);
            rv0 = *(const v8bf*)gVn; rv1 = *(const v8bf*)(gVn + 8);
            gKn += (long)32 * HE;
            gVn += 32;
        }
#endif
        const __bf16* Kt  = &Kl[cur][0];
        const __bf16* Vts = &Vl[cur][0];

        // ---- S = Q K^T: batch all 8 K fragments, then back-to-back WMMAs ----
        v16bf kf[8];
#pragma unroll
        for (int kk = 0; kk < 4; ++kk) {
            kf[2 * kk]     = load_b(Kt, KSTR, 0,  kk * 32, lane);
            kf[2 * kk + 1] = load_b(Kt, KSTR, 16, kk * 32, lane);
        }
        v8f s0 = {}, s1 = {};
#pragma unroll
        for (int kk = 0; kk < 4; ++kk) {
            s0 = wmma_bf16(qa[kk], kf[2 * kk],     s0);
            s1 = wmma_bf16(qa[kk], kf[2 * kk + 1], s1);
        }

        // ---- online softmax (DPP butterfly, no LDS) ----
        float alpha[8], rs[8];
#pragma unroll
        for (int r = 0; r < 8; ++r) {
            const float tmx = red16_max(fmaxf(s0[r], s1[r]));
            const float mn  = fmaxf(mrow[r], tmx);
            alpha[r] = __expf(mrow[r] - mn);
            mrow[r] = mn;
            const float p0 = __expf(s0[r] - mn);
            const float p1 = __expf(s1[r] - mn);
            s0[r] = p0; s1[r] = p1;
            rs[r] = red16_sum(p0 + p1);
        }
#pragma unroll
        for (int r = 0; r < 8; ++r) lrow[r] = lrow[r] * alpha[r] + rs[r];
#pragma unroll
        for (int e = 0; e < 8; ++e)
#pragma unroll
            for (int r = 0; r < 8; ++r) o[e][r] *= alpha[r];

        // ---- P: C-layout -> LDS (padded row-major) -> A-layout reload ----
#pragma unroll
        for (int r = 0; r < 8; ++r) {
            pl[(r + 8 * hl) * PSTR + n]      = (__bf16)s0[r];
            pl[(r + 8 * hl) * PSTR + 16 + n] = (__bf16)s1[r];
        }
        v16bf pf = load_a(pl, PSTR, 0, 0, lane);

        // ---- O += P V: batch all 8 V fragments, then back-to-back WMMAs ----
        v16bf vf[8];
#pragma unroll
        for (int e = 0; e < 8; ++e) vf[e] = load_b(Vts, VSTR, e * 16, 0, lane);
#pragma unroll
        for (int e = 0; e < 8; ++e) o[e] = wmma_bf16(pf, vf[e], o[e]);

#if HAVE_ASYNC_LDS
        wait_async0();                     // next tile's DMA landed
        __syncthreads();
#else
        __syncthreads();                   // this buffer free for reuse
#endif
    }

    // ---- epilogue ----
    float linv[8];
#pragma unroll
    for (int r = 0; r < 8; ++r) linv[r] = 1.0f / lrow[r];
    __bf16* orow = Op + (long)(b * T + t0) * HE + (long)h * E;
#pragma unroll
    for (int e = 0; e < 8; ++e)
#pragma unroll
        for (int r = 0; r < 8; ++r)
            orow[(long)(r + 8 * hl) * HE + e * 16 + n] =
                (__bf16)(o[e][r] * linv[r]);
}

// ---------------- output projection: O[M,HE] @ Wu[HE,E] (Wut stored [E,HE]) ----------------
__global__ __launch_bounds__(128) void final_gemm_kernel(
        const __bf16* __restrict__ Oc, const __bf16* __restrict__ Wut,
        float* __restrict__ Y) {
    const int lane = threadIdx.x & 31;
    const int hl = lane >> 4, n = lane & 15;
    const int gw = blockIdx.x * 4 + (threadIdx.x >> 5);
    const int mt = gw >> 3;
    const int nt = gw & 7;

    v8f acc = {};
#pragma unroll 4
    for (int kk = 0; kk < HE; kk += 32) {
        v16bf a = load_a(Oc, HE, mt * 16, kk, lane);
        v16bf b = load_b(Wut, HE, nt * 16, kk, lane);
        acc = wmma_bf16(a, b, acc);
    }
#pragma unroll
    for (int r = 0; r < 8; ++r)
        Y[(long)(mt * 16 + r + 8 * hl) * E + nt * 16 + n] = acc[r];
}

// ---------------- launcher ----------------
extern "C" void kernel_launch(void* const* d_in, const int* in_sizes, int n_in,
                              void* d_out, int out_size, void* d_ws, size_t ws_size,
                              hipStream_t stream) {
    // setup_inputs order: k, q, v, Wk, Wq, Wv, Wu
    const float* k_in = (const float*)d_in[0];
    const float* q_in = (const float*)d_in[1];
    const float* v_in = (const float*)d_in[2];
    const float* Wk   = (const float*)d_in[3];
    const float* Wq   = (const float*)d_in[4];
    const float* Wv   = (const float*)d_in[5];
    const float* Wu   = (const float*)d_in[6];
    float* out = (float*)d_out;

    char* ws = (char*)d_ws;
    size_t off = 0;
    auto alloc = [&](size_t bytes) -> char* {
        char* p = ws + off;
        off += (bytes + 255) & ~(size_t)255;
        return p;
    };
    __bf16* qb  = (__bf16*)alloc((size_t)M * E * 2);
    __bf16* kb  = (__bf16*)alloc((size_t)M * E * 2);
    __bf16* vb  = (__bf16*)alloc((size_t)M * E * 2);
    __bf16* Wqt = (__bf16*)alloc((size_t)HE * E * 2);
    __bf16* Wkt = (__bf16*)alloc((size_t)HE * E * 2);
    __bf16* Wvt = (__bf16*)alloc((size_t)HE * E * 2);
    __bf16* Wut = (__bf16*)alloc((size_t)E * HE * 2);
    __bf16* Qp  = (__bf16*)alloc((size_t)M * HE * 2);
    __bf16* Kp  = (__bf16*)alloc((size_t)M * HE * 2);
    __bf16* Vtb = (__bf16*)alloc((size_t)B * H * E * T * 2);
    __bf16* Op  = (__bf16*)alloc((size_t)M * HE * 2);

    const float qscale = 0.08838834764831845f;  // 1/sqrt(E) folded into Q

    const int nAct = M * E;
    cvt_bf16_kernel<<<nAct / 256, 256, 0, stream>>>(q_in, qb, nAct, qscale);
    cvt_bf16_kernel<<<nAct / 256, 256, 0, stream>>>(k_in, kb, nAct, 1.0f);
    cvt_bf16_kernel<<<nAct / 256, 256, 0, stream>>>(v_in, vb, nAct, 1.0f);

    const int nW = E * HE;
    tcvt_bf16_kernel<<<nW / 256, 256, 0, stream>>>(Wq, Wqt, E, HE);
    tcvt_bf16_kernel<<<nW / 256, 256, 0, stream>>>(Wk, Wkt, E, HE);
    tcvt_bf16_kernel<<<nW / 256, 256, 0, stream>>>(Wv, Wvt, E, HE);
    tcvt_bf16_kernel<<<nW / 256, 256, 0, stream>>>(Wu, Wut, HE, E);

    proj_gemm_kernel<0><<<8192, 128, 0, stream>>>(qb, Wqt, Qp);
    proj_gemm_kernel<0><<<8192, 128, 0, stream>>>(kb, Wkt, Kp);
    proj_gemm_kernel<1><<<8192, 128, 0, stream>>>(vb, Wvt, Vtb);

    // 512 blocks x 256 threads: (b,h) x 32 query blocks of 128 rows
    flash_attn_kernel<<<512, 256, 0, stream>>>(Qp, Kp, Vtb, Op);

    final_gemm_kernel<<<1024, 128, 0, stream>>>(Op, Wut, out);
}
